// lstmlayer_45578192945387
// MI455X (gfx1250) — matrix operand order repaired and checked by reading, not compiled, not run on hardware
//
#include <hip/hip_runtime.h>

// ---------------------------------------------------------------------------
// LSTM layer for MI455X (gfx1250): bf16 WMMA GEMMs + fused recurrent step.
//   B=64, S=512, D=1024, H4=4096
// Phase 0: convert x, Wx, Wh to bf16 (workspace); zero h0/c0.
// Phase 1: gx[B*S,4096] = x_bf16 @ Wx_bf16^T + bx  (v_wmma_f32_16x16x32_bf16,
//          software-pipelined double-buffered fragments)
// Phase 2: 512x lstm_step: async-to-LDS h staging, per-gate WMMA,
//          fused sigmoid/tanh cell update, ping-pong bf16 h state.
// ---------------------------------------------------------------------------

#define LSTM_B  64
#define LSTM_S  512
#define LSTM_D  1024
#define LSTM_H4 4096
#define LSTM_BS (LSTM_B * LSTM_S)   // 32768 flattened rows of x

typedef __attribute__((ext_vector_type(16))) __bf16 v16bf;
typedef __attribute__((ext_vector_type(8)))  float  v8f;

union Frag { v16bf v; uint4 u[2]; };   // 16 bf16 per lane (2 x 16B chunks)

__device__ __forceinline__ unsigned short f2bf(float f) {
    unsigned int u = __float_as_uint(f);
    unsigned int r = u + 0x7FFFu + ((u >> 16) & 1u);   // round-to-nearest-even
    return (unsigned short)(r >> 16);
}

__device__ __forceinline__ float sigm_f(float x) {
    return 1.0f / (1.0f + __expf(-x));
}
__device__ __forceinline__ float tanh_f(float x) {
    float e = __expf(-2.0f * x);
    return (1.0f - e) / (1.0f + e);
}

// ---------------------------------------------------------------- conversions
__global__ void __launch_bounds__(256)
f32_to_bf16_kernel(const float* __restrict__ src, unsigned short* __restrict__ dst, int n) {
    int i = blockIdx.x * 256 + threadIdx.x;
    if (i < n) dst[i] = f2bf(src[i]);
}

__global__ void __launch_bounds__(256)
init_state_kernel(unsigned short* __restrict__ h0, float* __restrict__ c0, int n) {
    int i = blockIdx.x * 256 + threadIdx.x;
    if (i < n) { h0[i] = 0; c0[i] = 0.0f; }
}

// ------------------------------------------------------------- gx = x Wx^T+bx
// grid = (H4/256, BS/16), block = 128 (4 waves). Wave w: 16x64 tile at
// n0 = bx*256 + w*64, m0 = by*16. K loop over D in steps of 32 with
// double-buffered fragments so loads for k+1 overlap WMMAs for k.
__global__ void __launch_bounds__(128)
gx_gemm_kernel(const unsigned short* __restrict__ Xb,
               const unsigned short* __restrict__ Wxb,
               const float* __restrict__ bx,
               float* __restrict__ gx) {
    const int lane  = threadIdx.x & 31;
    const int wave  = threadIdx.x >> 5;
    const int lm    = lane & 15;
    const int lhalf = lane >> 4;
    const int m0 = blockIdx.y << 4;
    const int n0 = (blockIdx.x << 8) + (wave << 6);

    const int akoff = lhalf * 8;     // A: elements {0..7,16..23} or {8..15,24..31}
    const int bkoff = lhalf * 8;     // B: elements {0..7,16..23} or {8..15,24..31}

    const unsigned short* arow = Xb + (size_t)(m0 + lm) * LSTM_D;
    const unsigned short* brow[4];
#pragma unroll
    for (int j = 0; j < 4; ++j)
        brow[j] = Wxb + (size_t)(n0 + j * 16 + lm) * LSTM_D;

    Frag a[2];
    Frag w[2][4];
    auto load_set = [&](int i, int kb) {
        a[i].u[0] = *(const uint4*)(arow + kb + akoff);
        a[i].u[1] = *(const uint4*)(arow + kb + akoff + 16);
#pragma unroll
        for (int j = 0; j < 4; ++j) {
            w[i][j].u[0] = *(const uint4*)(brow[j] + kb + bkoff);
            w[i][j].u[1] = *(const uint4*)(brow[j] + kb + bkoff + 16);
        }
    };

    v8f acc[4] = {v8f{}, v8f{}, v8f{}, v8f{}};
    load_set(0, 0);
#pragma unroll 4
    for (int kb = 0; kb < LSTM_D; kb += 32) {
        const int cur = (kb >> 5) & 1;
        const int nxt = cur ^ 1;
        if (kb + 32 < LSTM_D) {
            load_set(nxt, kb + 32);                      // overlap with WMMAs below
            __builtin_prefetch(arow + kb + 128 + akoff, 0, 1);   // global_prefetch_b8
            __builtin_prefetch(brow[0] + kb + 128 + bkoff, 0, 1);
        }
#pragma unroll
        for (int j = 0; j < 4; ++j)
            acc[j] = __builtin_amdgcn_wmma_f32_16x16x32_bf16(
                false, a[cur].v, false, w[cur][j].v, (short)0, acc[j], false, false);
    }

    // C/D layout: VGPR r -> row m0 + r + 8*lhalf, column = lm within 16-tile.
#pragma unroll
    for (int j = 0; j < 4; ++j) {
        const int gcol = n0 + j * 16 + lm;
        const float bias = bx[gcol];
#pragma unroll
        for (int r = 0; r < 8; ++r) {
            const int m = m0 + r + 8 * lhalf;
            gx[(size_t)m * LSTM_H4 + gcol] = acc[j][r] + bias;
        }
    }
}

// ------------------------------------------------------------ recurrent step
// grid = (D/64, B/16), block = 128 (4 waves, one per gate i/f/g/o).
// h(t-1) slice staged into LDS with GLOBAL_LOAD_ASYNC_TO_LDS_B128 (ASYNCcnt),
// per-gate 16x64 WMMA tile with double-buffered B fragments, tiles meet in
// LDS, fused activations + cell update. h ping-pongs between two buffers.
__global__ void __launch_bounds__(128)
lstm_step_kernel(const unsigned short* __restrict__ Whb,
                 const float* __restrict__ bh,
                 const float* __restrict__ gx,
                 const unsigned short* __restrict__ hin,
                 unsigned short* __restrict__ hout,
                 float* __restrict__ cstate,
                 float* __restrict__ out,
                 int t) {
    __shared__ unsigned short hA[16 * LSTM_D];   // 32 KB
    __shared__ float gsm[4][16][64];             // 16 KB

    const int tid   = threadIdx.x;
    const int lane  = tid & 31;
    const int wave  = tid >> 5;
    const int lm    = lane & 15;
    const int lhalf = lane >> 4;
    const int m0 = blockIdx.y << 4;   // batch base
    const int n0 = blockIdx.x << 6;   // hidden-dim base

    // ---- async-stage h(t-1) rows [m0, m0+16) into LDS, 16B per lane ----
    for (int idx = tid; idx < (16 * LSTM_D) / 8; idx += 128) {
        const int row  = idx >> 7;          // 128 uint4 per row
        const int col8 = (idx & 127) * 8;   // element offset
        // LDS flat address low 32 bits == LDS byte offset (ISA 10.2 aperture rule)
        unsigned dst_lds = (unsigned)(unsigned long long)(const void*)&hA[row * LSTM_D + col8];
        unsigned long long src_ga =
            (unsigned long long)(const void*)&hin[(size_t)(m0 + row) * LSTM_D + col8];
        asm volatile("global_load_async_to_lds_b128 %0, %1, off"
                     :: "v"(dst_lds), "v"(src_ga) : "memory");
    }
    asm volatile("s_wait_asynccnt 0x0" ::: "memory");
    __syncthreads();

    // ---- wave `wave` computes gate tile: h @ Wh^T over rows wave*1024+n0.. ----
    const int akoff = lhalf * 8;
    const int bkoff = lhalf * 8;
    const unsigned short* arow = &hA[lm * LSTM_D];
    const unsigned short* brow[4];
#pragma unroll
    for (int j = 0; j < 4; ++j)
        brow[j] = Whb + (size_t)(wave * LSTM_D + n0 + j * 16 + lm) * LSTM_D;

    Frag a;                 // LDS-sourced, low latency: single-buffered
    Frag w[2][4];           // global-sourced: double-buffered
    auto load_b = [&](int i, int kb) {
#pragma unroll
        for (int j = 0; j < 4; ++j) {
            w[i][j].u[0] = *(const uint4*)(brow[j] + kb + bkoff);
            w[i][j].u[1] = *(const uint4*)(brow[j] + kb + bkoff + 16);
        }
    };

    v8f acc[4] = {v8f{}, v8f{}, v8f{}, v8f{}};
    load_b(0, 0);
#pragma unroll 4
    for (int kb = 0; kb < LSTM_D; kb += 32) {
        const int cur = (kb >> 5) & 1;
        const int nxt = cur ^ 1;
        if (kb + 32 < LSTM_D) load_b(nxt, kb + 32);
        a.u[0] = *(const uint4*)(arow + kb + akoff);        // ds_load_b128
        a.u[1] = *(const uint4*)(arow + kb + akoff + 16);
#pragma unroll
        for (int j = 0; j < 4; ++j)
            acc[j] = __builtin_amdgcn_wmma_f32_16x16x32_bf16(
                false, a.v, false, w[cur][j].v, (short)0, acc[j], false, false);
    }

#pragma unroll
    for (int j = 0; j < 4; ++j)
#pragma unroll
        for (int r = 0; r < 8; ++r)
            gsm[wave][r + 8 * lhalf][j * 16 + lm] = acc[j][r];
    __syncthreads();

    // ---- fused gate activations + cell update (8 elements/thread) ----
    for (int idx = tid; idx < 16 * 64; idx += 128) {
        const int m = idx >> 6;
        const int n = idx & 63;
        const int b = m0 + m;
        const int d = n0 + n;
        const size_t gxb = ((size_t)(b * LSTM_S + t)) * LSTM_H4 + d;

        const float ri = gsm[0][m][n] + gx[gxb +    0] + bh[   0 + d];
        const float rf = gsm[1][m][n] + gx[gxb + 1024] + bh[1024 + d];
        const float rg = gsm[2][m][n] + gx[gxb + 2048] + bh[2048 + d];
        const float ro = gsm[3][m][n] + gx[gxb + 3072] + bh[3072 + d];

        const size_t sidx = (size_t)b * LSTM_D + d;
        const float c_old = cstate[sidx];
        const float c_new = sigm_f(rf) * c_old + sigm_f(ri) * tanh_f(rg);
        const float h_new = sigm_f(ro) * tanh_f(c_new);

        cstate[sidx] = c_new;
        hout[sidx]   = f2bf(h_new);

        const size_t oidx = ((size_t)(b * LSTM_S + t)) * LSTM_D + d;
        out[oidx] = h_new;                                        // hs
        out[(size_t)LSTM_B * LSTM_S * LSTM_D + oidx] = c_new;     // cs
    }
}

// ---------------------------------------------------------------------------
extern "C" void kernel_launch(void* const* d_in, const int* in_sizes, int n_in,
                              void* d_out, int out_size, void* d_ws, size_t ws_size,
                              hipStream_t stream) {
    (void)in_sizes; (void)n_in; (void)out_size; (void)ws_size;

    const float* x  = (const float*)d_in[0];   // [B,S,D]
    const float* Wx = (const float*)d_in[1];   // [4D,D]
    const float* bx = (const float*)d_in[2];   // [4D]
    const float* Wh = (const float*)d_in[3];   // [4D,D]
    const float* bh = (const float*)d_in[4];   // [4D]
    float* out = (float*)d_out;                // hs [B,S,D] ++ cs [B,S,D]

    char* ws = (char*)d_ws;
    const size_t X_ELEMS = (size_t)LSTM_BS * LSTM_D;     // 33,554,432
    const size_t W_ELEMS = (size_t)LSTM_H4 * LSTM_D;     //  4,194,304
    const size_t GX_BYTES = (size_t)LSTM_BS * LSTM_H4 * sizeof(float);
    const size_t H_BYTES  = (size_t)LSTM_B * LSTM_D * sizeof(unsigned short);
    const size_t C_BYTES  = (size_t)LSTM_B * LSTM_D * sizeof(float);

    size_t off = 0;
    unsigned short* Xb  = (unsigned short*)(ws + off); off += X_ELEMS * 2;
    unsigned short* Wxb = (unsigned short*)(ws + off); off += W_ELEMS * 2;
    unsigned short* Whb = (unsigned short*)(ws + off); off += W_ELEMS * 2;
    float*          gxp = (float*)(ws + off);          off += GX_BYTES;
    unsigned short* hb0 = (unsigned short*)(ws + off); off += H_BYTES;
    unsigned short* hb1 = (unsigned short*)(ws + off); off += H_BYTES;
    float*          cst = (float*)(ws + off);          off += C_BYTES;

    // Phase 0: conversions + state init (every call; deterministic).
    f32_to_bf16_kernel<<<dim3((unsigned)((X_ELEMS + 255) / 256)), 256, 0, stream>>>(x,  Xb,  (int)X_ELEMS);
    f32_to_bf16_kernel<<<dim3((unsigned)((W_ELEMS + 255) / 256)), 256, 0, stream>>>(Wx, Wxb, (int)W_ELEMS);
    f32_to_bf16_kernel<<<dim3((unsigned)((W_ELEMS + 255) / 256)), 256, 0, stream>>>(Wh, Whb, (int)W_ELEMS);
    init_state_kernel<<<dim3((LSTM_B * LSTM_D + 255) / 256), 256, 0, stream>>>(hb0, cst, LSTM_B * LSTM_D);

    // Phase 1: hoisted input projection (one big WMMA GEMM).
    gx_gemm_kernel<<<dim3(LSTM_H4 / 256, LSTM_BS / 16), 128, 0, stream>>>(Xb, Wxb, bx, gxp);

    // Phase 2: sequential scan, ping-ponging the bf16 hidden state.
    for (int t = 0; t < LSTM_S; ++t) {
        const unsigned short* hin  = (t & 1) ? hb1 : hb0;
        unsigned short*       hout = (t & 1) ? hb0 : hb1;
        lstm_step_kernel<<<dim3(LSTM_D / 64, LSTM_B / 16), 128, 0, stream>>>(
            Whb, bh, gxp, hin, hout, cst, out, t);
    }
}
